// CosineSimilarity_1314259992867
// MI455X (gfx1250) — matrix operand order repaired and checked by reading, not compiled, not run on hardware
//
#include <hip/hip_runtime.h>
#include <hip/hip_bf16.h>

// ---------------------------------------------------------------------------
// Problem constants (reference: tensor_1 [8192,256], tensor_2 [16384,256])
// ---------------------------------------------------------------------------
#define DIM_K   256
#define N_ROWS  8192
#define M_ROWS  16384
#define TILES_N (M_ROWS / 16)           // 1024 column tiles
#define WAVES   8                        // 256-thread block = 8 wave32
#define MT_PER_WAVE 2                    // M-tiles (16 rows each) per wave
#define ROWS_PER_BLOCK (WAVES * MT_PER_WAVE * 16)   // 256 rows
#define SPLITS  16                       // N-dimension splits
#define TILES_PER_SPLIT (TILES_N / SPLITS)          // 64 tiles per split

typedef _Float16 v16h __attribute__((ext_vector_type(16)));
typedef _Float16 v8h  __attribute__((ext_vector_type(8)));
typedef float    v8f  __attribute__((ext_vector_type(8)));

// ---------------------------------------------------------------------------
// Kernel 1: L2-normalize rows (f32 math) and emit f16 rows into workspace.
// One wave32 per row: 32 lanes x 8 elements = 256.
// ---------------------------------------------------------------------------
__global__ __launch_bounds__(256)
void normalize_f16(const float* __restrict__ in, _Float16* __restrict__ out, int rows) {
    const int wave = threadIdx.x >> 5;
    const int lane = threadIdx.x & 31;
    const int row  = blockIdx.x * WAVES + wave;
    if (row >= rows) return;

    const float4* rp = reinterpret_cast<const float4*>(in + (size_t)row * DIM_K);
    float4 a = rp[lane * 2 + 0];
    float4 b = rp[lane * 2 + 1];

    float s = a.x * a.x + a.y * a.y + a.z * a.z + a.w * a.w +
              b.x * b.x + b.y * b.y + b.z * b.z + b.w * b.w;
    #pragma unroll
    for (int off = 16; off > 0; off >>= 1)
        s += __shfl_xor(s, off, 32);

    const float r = rsqrtf(s);

    v8h o;
    o[0] = (_Float16)(a.x * r); o[1] = (_Float16)(a.y * r);
    o[2] = (_Float16)(a.z * r); o[3] = (_Float16)(a.w * r);
    o[4] = (_Float16)(b.x * r); o[5] = (_Float16)(b.y * r);
    o[6] = (_Float16)(b.z * r); o[7] = (_Float16)(b.w * r);
    *reinterpret_cast<v8h*>(out + (size_t)row * DIM_K + lane * 8) = o;
}

// ---------------------------------------------------------------------------
// Branchless top-3 insertion, 4 VALU ops, exploiting invariant t0 >= t1 >= t2:
//   t0' = max(t0, v)
//   t1' = med3(t0, t1, v)          (second-largest of {t0,t1,v} since t0>=t1)
//   t2' = max(t2, min(t1, v))      (min3(t0,t1,v) == min(t1,v) since t0>=t1)
// ---------------------------------------------------------------------------
__device__ __forceinline__ void top3_insert(float v, float& t0, float& t1, float& t2) {
    float n0 = fmaxf(t0, v);
    float n1 = __builtin_amdgcn_fmed3f(t0, t1, v);
    float n2 = fmaxf(t2, fminf(t1, v));
    t0 = n0; t1 = n1; t2 = n2;
}

// ---------------------------------------------------------------------------
// One 16-column tile: 8 K-steps x 2 M-tiles = 16 WMMAs on two independent
// accumulator chains, B fragments loaded as contiguous 32B per lane.
// ---------------------------------------------------------------------------
__device__ __forceinline__ void compute_tile(const _Float16* __restrict__ bRow, int hg,
                                             const v16h aReg[MT_PER_WAVE][8],
                                             v8f& out0, v8f& out1) {
    v8f c0 = {};
    v8f c1 = {};
    #pragma unroll
    for (int kt = 0; kt < 8; ++kt) {
        const int kbase = kt * 32 + hg * 16;
        v16h b = *reinterpret_cast<const v16h*>(bRow + kbase);   // 32B contiguous
        c0 = __builtin_amdgcn_wmma_f32_16x16x32_f16(
                 false, aReg[0][kt], false, b, (short)0, c0, false, false);
        c1 = __builtin_amdgcn_wmma_f32_16x16x32_f16(
                 false, aReg[1][kt], false, b, (short)0, c1, false, false);
    }
    out0 = c0;
    out1 = c1;
}

// ---------------------------------------------------------------------------
// Kernel 2: block = 256 rows of A x one N-split (64 column tiles).
// All 8 waves walk the SAME nt sequence (B loads broadcast through WGP$);
// each wave owns 2 M-tiles. Top-3 selection is software-pipelined one tile
// behind the WMMAs so the selection VALU fills the XDL shadow with no
// WMMA->VALU hazard stalls.
//
// WMMA 16x16x32 f16 VGPR layouts (ISA 7.12.2, wave32):
//   A (16x32 MxK): lane L holds row M = L%16; halves 0..7 -> k = h + 8*(L/16),
//                  halves 8..15 -> k = 16 + (h-8) + 8*(L/16).
//   B (32x16 KxN): lane L holds col N = L%16; this code loads A and B alike: halves 0..15 -> k = h + 16*(L/16).
//   C/D f32:       lane L, VGPR r -> element [M = r + 8*(L/16)][N = L%16].
// ---------------------------------------------------------------------------
__global__ __launch_bounds__(256)
void cosine_top3_part(const _Float16* __restrict__ A, const _Float16* __restrict__ B,
                      float* __restrict__ cand_out) {
    __shared__ float sc[WAVES][MT_PER_WAVE][16][16][3];   // 48 KB

    const int tid  = threadIdx.x;
    const int wave = tid >> 5;
    const int lane = tid & 31;
    const int l16  = lane & 15;
    const int hg   = lane >> 4;          // half-wave group (0 or 1)
    const int mBlk  = blockIdx.x;        // 0..31 (256-row blocks)
    const int split = blockIdx.y;        // 0..15 (N splits)

    // --- Load both M-tiles' full-K A operands into registers ---
    v16h aReg[MT_PER_WAVE][8];
    #pragma unroll
    for (int t = 0; t < MT_PER_WAVE; ++t) {
        const _Float16* aRow =
            A + (size_t)(mBlk * ROWS_PER_BLOCK + wave * (MT_PER_WAVE * 16) + t * 16 + l16) * DIM_K;
        #pragma unroll
        for (int kt = 0; kt < 8; ++kt) {
            const int kbase = kt * 32 + hg * 16;
            v8h lo = *reinterpret_cast<const v8h*>(aRow + kbase);        // k = 16*hg + 0..7
            v8h hi = *reinterpret_cast<const v8h*>(aRow + kbase + 8);    // k = 16*hg + 8..15
            #pragma unroll
            for (int i = 0; i < 8; ++i) {
                aReg[t][kt][i] = lo[i];
                aReg[t][kt][i + 8] = hi[i];
            }
        }
    }

    // Per-lane running top-3 for each of 2 tiles x 8 accumulator rows.
    float t0[MT_PER_WAVE][8], t1[MT_PER_WAVE][8], t2[MT_PER_WAVE][8];
    #pragma unroll
    for (int t = 0; t < MT_PER_WAVE; ++t)
        #pragma unroll
        for (int r = 0; r < 8; ++r) { t0[t][r] = -2.0f; t1[t][r] = -2.0f; t2[t][r] = -2.0f; }

    const int ntBase = split * TILES_PER_SPLIT;
    const _Float16* bBase = B + (size_t)l16 * DIM_K;

    // --- Peel tile 0, then pipeline: WMMAs of tile i overlap inserts of i-1 ---
    v8f p0, p1;
    compute_tile(bBase + (size_t)(ntBase) * (16 * DIM_K), hg, aReg, p0, p1);

    #pragma unroll 1
    for (int i = 1; i < TILES_PER_SPLIT; ++i) {
        // Prefetch tile i+1 (clamped, branchless): 32 lanes x 2 x 128B = 8KB tile.
        {
            const int ip = (i + 1 < TILES_PER_SPLIT) ? (i + 1) : i;
            const char* nxt = (const char*)(bBase + (size_t)(ntBase + ip) * (16 * DIM_K));
            __builtin_prefetch(nxt + hg * 256, 0, 3);
            __builtin_prefetch(nxt + hg * 256 + 128, 0, 3);
        }

        v8f c0, c1;
        compute_tile(bBase + (size_t)(ntBase + i) * (16 * DIM_K), hg, aReg, c0, c1);

        // Insert PREVIOUS tile's results: independent of in-flight WMMAs.
        #pragma unroll
        for (int r = 0; r < 8; ++r) {
            top3_insert(p0[r], t0[0][r], t1[0][r], t2[0][r]);
            top3_insert(p1[r], t0[1][r], t1[1][r], t2[1][r]);
        }
        p0 = c0;
        p1 = c1;
    }
    // Drain the last tile's results.
    #pragma unroll
    for (int r = 0; r < 8; ++r) {
        top3_insert(p0[r], t0[0][r], t1[0][r], t2[0][r]);
        top3_insert(p1[r], t0[1][r], t1[1][r], t2[1][r]);
    }

    // --- Stash per-lane candidates: tile t, row m = r + 8*hg, lane slot l16 ---
    #pragma unroll
    for (int t = 0; t < MT_PER_WAVE; ++t)
        #pragma unroll
        for (int r = 0; r < 8; ++r) {
            const int m = r + 8 * hg;
            sc[wave][t][m][l16][0] = t0[t][r];
            sc[wave][t][m][l16][1] = t1[t][r];
            sc[wave][t][m][l16][2] = t2[t][r];
        }
    __syncthreads();

    // --- One thread per row-in-block (256 rows): merge 16 lanes x 3 candidates ---
    {
        const int w = tid >> 5;
        const int t = (tid >> 4) & 1;
        const int m = tid & 15;
        float s0 = -2.0f, s1 = -2.0f, s2 = -2.0f;
        #pragma unroll
        for (int l = 0; l < 16; ++l) {
            top3_insert(sc[w][t][m][l][0], s0, s1, s2);
            top3_insert(sc[w][t][m][l][1], s0, s1, s2);
            top3_insert(sc[w][t][m][l][2], s0, s1, s2);
        }
        const size_t row = (size_t)mBlk * ROWS_PER_BLOCK + tid;
        float* p = cand_out + row * (SPLITS * 3) + split * 3;
        p[0] = s0; p[1] = s1; p[2] = s2;
    }
}

// ---------------------------------------------------------------------------
// Kernel 3: merge per-split candidates (16 splits x 3) -> mean of global top-3.
// ---------------------------------------------------------------------------
__global__ __launch_bounds__(256)
void final_merge(const float* __restrict__ cand, float* __restrict__ out) {
    const int row = blockIdx.x * 256 + threadIdx.x;
    if (row >= N_ROWS) return;
    const float* p = cand + (size_t)row * (SPLITS * 3);
    float s0 = -2.0f, s1 = -2.0f, s2 = -2.0f;
    #pragma unroll
    for (int i = 0; i < SPLITS * 3; ++i)
        top3_insert(p[i], s0, s1, s2);
    out[row] = (s0 + s1 + s2) * (1.0f / 3.0f);
}

// ---------------------------------------------------------------------------
// Launcher (graph-capture safe). Workspace layout:
//   [0, 4MB)      n1 f16    (8192*256*2)
//   [4MB, 12MB)   n2 f16    (16384*256*2)
//   [12MB, ...)   candidates f32 (8192 * 16 * 3 * 4 = 1.5MB)
// ---------------------------------------------------------------------------
extern "C" void kernel_launch(void* const* d_in, const int* in_sizes, int n_in,
                              void* d_out, int out_size, void* d_ws, size_t ws_size,
                              hipStream_t stream) {
    const float* t1 = (const float*)d_in[0];   // [8192, 256]
    const float* t2 = (const float*)d_in[1];   // [16384, 256]
    float* out = (float*)d_out;                // [8192]

    _Float16* n1h  = (_Float16*)d_ws;
    _Float16* n2h  = n1h + (size_t)N_ROWS * DIM_K;
    float*    cand = (float*)(n2h + (size_t)M_ROWS * DIM_K);

    normalize_f16<<<N_ROWS / WAVES, 256, 0, stream>>>(t1, n1h, N_ROWS);
    normalize_f16<<<M_ROWS / WAVES, 256, 0, stream>>>(t2, n2h, M_ROWS);

    dim3 grid(N_ROWS / ROWS_PER_BLOCK, SPLITS);   // (32, 16) = 512 blocks
    cosine_top3_part<<<grid, 256, 0, stream>>>(n1h, n2h, cand);

    final_merge<<<N_ROWS / 256, 256, 0, stream>>>(cand, out);
}
